// SelfAttention_7121055777010
// MI455X (gfx1250) — matrix operand order repaired and checked by reading, not compiled, not run on hardware
//
#include <hip/hip_runtime.h>
#include <hip/hip_bf16.h>
#include <math.h>

#define BN   8
#define CC   256
#define NN   4096
#define CQ   32

#define XS_STRIDE 264   // x-tile LDS row stride (halves), padded vs 256
#define PS_STRIDE 40    // P-tile LDS row stride (halves), padded vs 32
#define TS_STRIDE 20    // out-transpose LDS row stride (floats), padded vs 16
#define KV_STRIDE 40    // K/V chunk LDS row stride (halves), padded vs 32 (TDM pad)

typedef __attribute__((ext_vector_type(16))) _Float16 v16h;
typedef __attribute__((ext_vector_type(8)))  _Float16 v8h;
typedef __attribute__((ext_vector_type(8)))  float    v8f;
typedef __attribute__((ext_vector_type(4)))  float    v4f;

typedef unsigned int u32x4 __attribute__((ext_vector_type(4)));
typedef int          i32x4 __attribute__((ext_vector_type(4)));
typedef int          i32x8 __attribute__((ext_vector_type(8)));

#if __has_builtin(__builtin_amdgcn_tensor_load_to_lds) && \
    __has_builtin(__builtin_amdgcn_s_wait_tensorcnt)
#define HAS_TDM 1
#else
#define HAS_TDM 0
#endif

__device__ __forceinline__ v16h cat16(v8h lo, v8h hi) {
    v16h r;
#pragma unroll
    for (int i = 0; i < 8; ++i) { r[i] = lo[i]; r[i + 8] = hi[i]; }
    return r;
}

__device__ __forceinline__ v8f wmma_f16(v16h a, v16h b, v8f c) {
    // D = A(16x32 f16) * B(32x16 f16) + C(16x16 f32)
    return __builtin_amdgcn_wmma_f32_16x16x32_f16(false, a, false, b, (short)0, c, false, false);
}

#if HAS_TDM
// Issue a 2-D TDM load: tile (tile_d0 x tile_d1) of f16 from a row-major
// tensor (row stride stride0, in elements) into LDS at lds_byte_addr, with
// optional LDS row padding. D# bit layout per CDNA5 ISA §8.3/8.4.
// This toolchain's builtin takes 6 args: (g0, g1, g2, g3, g4, cpol).
__device__ __forceinline__ void tdm_load_2d(unsigned lds_byte_addr,
                                            const _Float16* gptr,
                                            unsigned tensor_d0, unsigned tensor_d1,
                                            unsigned long long stride0,
                                            unsigned tile_d0, unsigned tile_d1,
                                            unsigned pad_en, unsigned pad_int,
                                            unsigned pad_amt) {
    unsigned long long ga = (unsigned long long)(size_t)gptr;
    u32x4 g0;
    g0[0] = 1u;                                                  // count=1, user D#
    g0[1] = lds_byte_addr;                                       // lds_addr
    g0[2] = (unsigned)(ga & 0xffffffffu);                        // global_addr[31:0]
    g0[3] = (unsigned)((ga >> 32) & 0x01ffffffu) | (2u << 30);   // global_addr[56:32], type=2
    i32x8 g1;
    g1[0] = (int)((1u << 16) |                                   // data_size = 2 bytes
                  (pad_en << 20) | (pad_int << 22) | (pad_amt << 25));
    g1[1] = (int)((tensor_d0 & 0xffffu) << 16);                  // tensor_dim0[15:0]
    g1[2] = (int)(((tensor_d0 >> 16) & 0xffffu) |                // tensor_dim0[31:16]
                  ((tensor_d1 & 0xffffu) << 16));                // tensor_dim1[15:0]
    g1[3] = (int)(((tensor_d1 >> 16) & 0xffffu) |                // tensor_dim1[31:16]
                  ((tile_d0 & 0xffffu) << 16));                  // tile_dim0
    g1[4] = (int)(tile_d1 & 0xffffu);                            // tile_dim1 (tile_dim2=0)
    g1[5] = (int)(unsigned)(stride0 & 0xffffffffu);              // tensor_dim0_stride[31:0]
    g1[6] = (int)(unsigned)((stride0 >> 32) & 0xffffu);          // stride[47:32] (dim1_stride=0)
    g1[7] = 0;
    i32x4 z4 = {0, 0, 0, 0};
    i32x8 z8 = {0, 0, 0, 0, 0, 0, 0, 0};
    __builtin_amdgcn_tensor_load_to_lds(g0, g1, z4, z4, z8, 0);
}
#endif

// ---------------------------------------------------------------------------
// Kernel 0: convert weights f32 -> f16 in workspace
// ---------------------------------------------------------------------------
__global__ void convert_weights(const float* __restrict__ wq,
                                const float* __restrict__ wk,
                                const float* __restrict__ wv,
                                _Float16* __restrict__ wq16,
                                _Float16* __restrict__ wk16,
                                _Float16* __restrict__ wv16) {
    int t = blockIdx.x * blockDim.x + threadIdx.x;
    int stride = gridDim.x * blockDim.x;
    const int nq = CQ * CC;
    const int nv = CC * CC;
    for (int i = t; i < nq; i += stride) wq16[i] = (_Float16)wq[i];
    for (int i = t; i < nq; i += stride) wk16[i] = (_Float16)wk[i];
    for (int i = t; i < nv; i += stride) wv16[i] = (_Float16)wv[i];
}

// ---------------------------------------------------------------------------
// Kernel 1: QKV projection with WMMA.
//   Q16/K16: [B][N][32] f16 (n-major)   V16: [B][C][N] f16 (c-major)
// ---------------------------------------------------------------------------
__global__ __launch_bounds__(256) void qkv_proj(
    const float* __restrict__ x,
    const _Float16* __restrict__ wq16,
    const _Float16* __restrict__ wk16,
    const _Float16* __restrict__ wv16,
    const float* __restrict__ bq,
    const float* __restrict__ bk,
    const float* __restrict__ bv,
    _Float16* __restrict__ Q16,
    _Float16* __restrict__ K16,
    _Float16* __restrict__ V16) {
    __shared__ _Float16 xs[16 * XS_STRIDE];        // [n][c] f16, padded

    const int b  = blockIdx.x / (NN / 16);
    const int n0 = (blockIdx.x % (NN / 16)) * 16;
    const int t  = threadIdx.x;

    const float* xb = x + (size_t)b * CC * NN;
#pragma unroll
    for (int p = 0; p < 16; ++p) {
        int c = p * 16 + (t >> 4);
        int n = t & 15;
        xs[n * XS_STRIDE + c] = (_Float16)xb[(size_t)c * NN + n0 + n];
    }
    __syncthreads();

    const int wave = t >> 5, lane = t & 31;
    const int hi = lane >> 4, lo16 = lane & 15;

    // A fragments invariant across output tiles: preload all 8 k-steps
    v16h afr[8];
#pragma unroll
    for (int kk = 0; kk < 8; ++kk) {
        const int cb = kk * 32;
        v8h alo = *(const v8h*)&xs[lo16 * XS_STRIDE + cb + 8 * hi];
        v8h ahi = *(const v8h*)&xs[lo16 * XS_STRIDE + cb + 16 + 8 * hi];
        afr[kk] = cat16(alo, ahi);
    }

    for (int tile = wave; tile < 20; tile += 8) {
        const _Float16* wsel;
        const float* bsel;
        int o0, kind;
        if (tile < 2)      { wsel = wq16; bsel = bq; o0 = tile * 16;        kind = 0; }
        else if (tile < 4) { wsel = wk16; bsel = bk; o0 = (tile - 2) * 16;  kind = 1; }
        else               { wsel = wv16; bsel = bv; o0 = (tile - 4) * 16;  kind = 2; }

        // preload all 8 B fragments; sched_barrier keeps them as one clause
        v16h bfr[8];
#pragma unroll
        for (int kk = 0; kk < 8; ++kk) {
            const _Float16* wrow = wsel + (size_t)(o0 + lo16) * CC + kk * 32 + 8 * hi;
            bfr[kk] = cat16(*(const v8h*)wrow, *(const v8h*)(wrow + 16));
        }
#if __has_builtin(__builtin_amdgcn_sched_barrier)
        __builtin_amdgcn_sched_barrier(0);
#endif
        v8f acc = {};
#pragma unroll
        for (int kk = 0; kk < 8; ++kk) acc = wmma_f16(afr[kk], bfr[kk], acc);

        const float bias = bsel[o0 + lo16];
        if (kind == 2) {
            v8h pk;
#pragma unroll
            for (int r = 0; r < 8; ++r) pk[r] = (_Float16)(acc[r] + bias);
            _Float16* dv = V16 + (size_t)b * CC * NN + (size_t)(o0 + lo16) * NN + n0 + 8 * hi;
            *(v8h*)dv = pk;
        } else {
            _Float16* dst = (kind == 0 ? Q16 : K16) + (size_t)b * NN * CQ;
#pragma unroll
            for (int r = 0; r < 8; ++r)
                dst[(size_t)(n0 + 8 * hi + r) * CQ + o0 + lo16] = (_Float16)(acc[r] + bias);
        }
    }
}

// ---------------------------------------------------------------------------
// Kernel 2: flash attention + residual.
// K/V chunks (32 keys) are DMA'd into double-buffered LDS by the Tensor Data
// Mover (wave 0 issues, TENSORcnt + workgroup barrier synchronize), with TDM
// LDS-padding giving conflict-free 40-half rows. All 8 waves consume the
// shared LDS chunk; per-wave state: 16 query rows, 256-wide f32 O accumulator.
// ---------------------------------------------------------------------------
__global__ __launch_bounds__(256) void flash_attn(
    const float* __restrict__ x,
    const _Float16* __restrict__ Q16,
    const _Float16* __restrict__ K16,
    const _Float16* __restrict__ V16,
    float* __restrict__ out) {
    __shared__ _Float16 Kls[2][32 * KV_STRIDE];    // 2 x 2.5 KB
    __shared__ _Float16 Vls[2][CC * KV_STRIDE];    // 2 x 20 KB
    __shared__ _Float16 Pls[8][16 * PS_STRIDE];    // per-wave P tile
    __shared__ float    Tls[8][16 * TS_STRIDE];    // per-wave out transpose

    const int b    = blockIdx.x >> 5;
    const int qt   = blockIdx.x & 31;
    const int t    = threadIdx.x;
    const int wave = t >> 5, lane = t & 31;
    const int hi = lane >> 4, lo16 = lane & 15;
    const int n0 = qt * 128 + wave * 16;

    const _Float16* Qb = Q16 + (size_t)b * NN * CQ;
    const _Float16* Kb = K16 + (size_t)b * NN * CQ;
    const _Float16* Vb = V16 + (size_t)b * CC * NN;

    // Q A-fragment (reused for all 128 key chunks)
    v8h qlo = *(const v8h*)&Qb[(size_t)(n0 + lo16) * CQ + 8 * hi];
    v8h qhi = *(const v8h*)&Qb[(size_t)(n0 + lo16) * CQ + 16 + 8 * hi];
    v16h qa = cat16(qlo, qhi);

    v8f o_acc[16];
#pragma unroll
    for (int ct = 0; ct < 16; ++ct) o_acc[ct] = (v8f){};
    float m_run[8], s_run[8];
#pragma unroll
    for (int r = 0; r < 8; ++r) { m_run[r] = -INFINITY; s_run[r] = 0.0f; }

    _Float16* P = Pls[wave];

#if HAS_TDM
    // prologue: DMA chunk 0 into buffer 0
    if (wave == 0) {
        tdm_load_2d((unsigned)(size_t)&Kls[0][0], Kb, CQ, NN, CQ, 32, 32, 1, 3, 3);
        tdm_load_2d((unsigned)(size_t)&Vls[0][0], Vb, NN, CC, NN, 32, CC, 1, 3, 3);
    }
#endif

    for (int m0 = 0; m0 < NN; m0 += 32) {
#if HAS_TDM
        const int cur = (m0 >> 5) & 1;
        if (wave == 0) __builtin_amdgcn_s_wait_tensorcnt(0);
        __syncthreads();                 // chunk `cur` ready; other buffer free
        if (wave == 0 && (m0 + 32) < NN) {
            tdm_load_2d((unsigned)(size_t)&Kls[cur ^ 1][0],
                        Kb + (size_t)(m0 + 32) * CQ, CQ, NN, CQ, 32, 32, 1, 3, 3);
            tdm_load_2d((unsigned)(size_t)&Vls[cur ^ 1][0],
                        Vb + (m0 + 32), NN, CC, NN, 32, CC, 1, 3, 3);
        }
#else
        const int cur = 0;
        __syncthreads();
        // cooperative fallback staging (global -> VGPR -> LDS)
        if (t < 128) {
            int row = t >> 2, seg = t & 3;
            v8h d = *(const v8h*)(Kb + (size_t)(m0 + row) * CQ + seg * 8);
            *(v8h*)&Kls[0][row * KV_STRIDE + seg * 8] = d;
        }
#pragma unroll
        for (int i = 0; i < 4; ++i) {
            int idx = i * 256 + t;
            int row = idx >> 2, seg = idx & 3;
            v8h d = *(const v8h*)(Vb + (size_t)row * NN + m0 + seg * 8);
            *(v8h*)&Vls[0][row * KV_STRIDE + seg * 8] = d;
        }
        __syncthreads();
#endif
        const _Float16* Kc = Kls[cur];
        const _Float16* Vc = Vls[cur];

        // ---- S = Q K^T : two 16x16 tiles from LDS ----
        v8f z = {};
        const _Float16* kr0 = Kc + lo16 * KV_STRIDE + 8 * hi;
        v16h kf0 = cat16(*(const v8h*)kr0, *(const v8h*)(kr0 + 16));
        v8f s0 = wmma_f16(qa, kf0, z);
        const _Float16* kr1 = Kc + (16 + lo16) * KV_STRIDE + 8 * hi;
        v16h kf1 = cat16(*(const v8h*)kr1, *(const v8h*)(kr1 + 16));
        v8f s1 = wmma_f16(qa, kf1, z);

        // ---- online softmax over this 32-wide chunk ----
        float alpha[8];
#pragma unroll
        for (int r = 0; r < 8; ++r) {
            float mx = fmaxf(s0[r], s1[r]);
            mx = fmaxf(mx, __shfl_xor(mx, 1, 32));
            mx = fmaxf(mx, __shfl_xor(mx, 2, 32));
            mx = fmaxf(mx, __shfl_xor(mx, 4, 32));
            mx = fmaxf(mx, __shfl_xor(mx, 8, 32));
            float mnew = fmaxf(m_run[r], mx);
            float a  = __expf(m_run[r] - mnew);
            float p0 = __expf(s0[r] - mnew);
            float p1 = __expf(s1[r] - mnew);
            s0[r] = p0; s1[r] = p1;
            float rs = p0 + p1;
            rs += __shfl_xor(rs, 1, 32);
            rs += __shfl_xor(rs, 2, 32);
            rs += __shfl_xor(rs, 4, 32);
            rs += __shfl_xor(rs, 8, 32);
            s_run[r] = s_run[r] * a + rs;
            m_run[r] = mnew;
            alpha[r] = a;
        }
#pragma unroll
        for (int ct = 0; ct < 16; ++ct)
#pragma unroll
            for (int r = 0; r < 8; ++r) o_acc[ct][r] *= alpha[r];

        // ---- transpose P (D-layout -> A-layout) through per-wave LDS ----
#pragma unroll
        for (int r = 0; r < 8; ++r) {
            P[(8 * hi + r) * PS_STRIDE + lo16]      = (_Float16)s0[r];
            P[(8 * hi + r) * PS_STRIDE + 16 + lo16] = (_Float16)s1[r];
        }
        v8h plo = *(const v8h*)&P[lo16 * PS_STRIDE + 8 * hi];
        v8h phi = *(const v8h*)&P[lo16 * PS_STRIDE + 16 + 8 * hi];
        v16h pa = cat16(plo, phi);

        // ---- O += P * V : 16 col-tiles, V fragments from shared LDS ----
#pragma unroll
        for (int ct = 0; ct < 16; ++ct) {
            const _Float16* vr = Vc + (ct * 16 + lo16) * KV_STRIDE + 8 * hi;
            v16h vf = cat16(*(const v8h*)vr, *(const v8h*)(vr + 16));
            o_acc[ct] = wmma_f16(pa, vf, o_acc[ct]);
        }
    }

    // ---- finalize: O / s + residual x, streaming NT writeback ----
    float inv[8];
#pragma unroll
    for (int r = 0; r < 8; ++r) inv[r] = 1.0f / s_run[r];

    float* T = Tls[wave];
    const int cl = lane >> 1, nh = (lane & 1) * 8;
    for (int ct = 0; ct < 16; ++ct) {
#pragma unroll
        for (int r = 0; r < 8; ++r)
            T[lo16 * TS_STRIDE + 8 * hi + r] = o_acc[ct][r] * inv[r];
        size_t base = ((size_t)b * CC + ct * 16 + cl) * NN + n0 + nh;
        const v4f* xr = (const v4f*)&x[base];
        v4f x0v = __builtin_nontemporal_load(xr);
        v4f x1v = __builtin_nontemporal_load(xr + 1);
        v4f t0v = *(const v4f*)&T[cl * TS_STRIDE + nh];
        v4f t1v = *(const v4f*)&T[cl * TS_STRIDE + nh + 4];
        v4f o0v = t0v + x0v;
        v4f o1v = t1v + x1v;
        v4f* orow = (v4f*)&out[base];
        __builtin_nontemporal_store(o0v, orow);
        __builtin_nontemporal_store(o1v, orow + 1);
    }
}

// ---------------------------------------------------------------------------
extern "C" void kernel_launch(void* const* d_in, const int* in_sizes, int n_in,
                              void* d_out, int out_size, void* d_ws, size_t ws_size,
                              hipStream_t stream) {
    (void)in_sizes; (void)n_in; (void)out_size; (void)ws_size;
    const float* x  = (const float*)d_in[0];
    const float* wq = (const float*)d_in[1];
    const float* bq = (const float*)d_in[2];
    const float* wk = (const float*)d_in[3];
    const float* bk = (const float*)d_in[4];
    const float* wv = (const float*)d_in[5];
    const float* bv = (const float*)d_in[6];
    float* out = (float*)d_out;

    _Float16* wq16 = (_Float16*)d_ws;
    _Float16* wk16 = wq16 + CQ * CC;
    _Float16* wv16 = wk16 + CQ * CC;
    _Float16* Q16  = wv16 + CC * CC;
    _Float16* K16  = Q16 + (size_t)BN * NN * CQ;
    _Float16* V16  = K16 + (size_t)BN * NN * CQ;

    convert_weights<<<64, 256, 0, stream>>>(wq, wk, wv, wq16, wk16, wv16);
    qkv_proj<<<BN * (NN / 16), 256, 0, stream>>>(x, wq16, wk16, wv16,
                                                 bq, bk, bv, Q16, K16, V16);
    flash_attn<<<BN * (NN / 128), 256, 0, stream>>>(x, Q16, K16, V16, out);
}
